// MPNNPredictor_38010460570544
// MI455X (gfx1250) — compile-verified, matrix operand order checked by reading, not run on hardware
//
#include <hip/hip_runtime.h>

#define DV 64
#define VN 4000
#define EN 40000
#define GN 40
#define NODE_IN 74
#define EDGE_IN 128
#define EDGE_HID 128
#define NSTEPS 6
#define S2S_ITERS 6

typedef __attribute__((ext_vector_type(16))) __bf16 v16bf;
typedef __attribute__((ext_vector_type(8)))  float  v8f;
typedef __attribute__((ext_vector_type(4)))  unsigned int u32x4;

__device__ __forceinline__ unsigned short f2bf(float f) {
  unsigned int u = __float_as_uint(f);
  u += 0x7fffu + ((u >> 16) & 1u);          // round-to-nearest-even
  return (unsigned short)(u >> 16);
}
__device__ __forceinline__ float bf2f(unsigned short b) {
  return __uint_as_float(((unsigned int)b) << 16);
}
__device__ __forceinline__ float sigmoidf(float x) { return 1.0f / (1.0f + expf(-x)); }

// ---------------- conversion kernels ----------------
__global__ void k_cvt_bf16(const float* __restrict__ in, unsigned short* __restrict__ out, int n) {
  int i = blockIdx.x * blockDim.x + threadIdx.x;
  if (i < n) out[i] = f2bf(in[i]);
}

// W (K x N) f32 row-major -> Wt (N x K) bf16 row-major
__global__ void k_cvt_transpose(const float* __restrict__ W, unsigned short* __restrict__ Wt,
                                int K, int N) {
  int i = blockIdx.x * blockDim.x + threadIdx.x;
  if (i >= K * N) return;
  int k = i / N, n = i - k * N;
  Wt[(size_t)n * K + k] = f2bf(W[i]);
}

__global__ void k_fill0(float* __restrict__ p, int n) {
  int i = blockIdx.x * blockDim.x + threadIdx.x;
  if (i < n) p[i] = 0.0f;
}

// ---------------- node projection: h = relu(nf @ Wp + bp) ----------------
__global__ void k_proj(const float* __restrict__ nf, const float* __restrict__ Wp,
                       const float* __restrict__ bp, float* __restrict__ hid,
                       unsigned short* __restrict__ hbf) {
  __shared__ float row[NODE_IN];
  int v = blockIdx.x, d = threadIdx.x;                 // blockDim = 64
  for (int k = d; k < NODE_IN; k += DV) row[k] = nf[v * NODE_IN + k];
  __syncthreads();
  float acc = bp[d];
  #pragma unroll
  for (int k = 0; k < NODE_IN; ++k) acc = fmaf(row[k], Wp[k * DV + d], acc);
  acc = fmaxf(acc, 0.0f);
  hid[(size_t)v * DV + d] = acc;
  hbf[(size_t)v * DV + d] = f2bf(acc);
}

// ---------------- bf16 WMMA GEMM: C = act(A @ Bt^T + bias), bf16 out ----------------
// A: M x 128 bf16 row-major (M multiple of 16). Bt: N x 128 bf16 row-major (pre-transposed
// weights so every fragment is two contiguous 16B loads). One wave owns a 16x64 tile.
// K = 128 is fully unrolled: ALL 40 b128 fragment loads are issued before the 16 WMMAs,
// so the compiler can use partial loadcnt waits and overlap the load tail with v_wmma.
union FragBF { u32x4 u[2]; v16bf v; };

template <int RELU>
__global__ void k_gemm_bf16(const unsigned short* __restrict__ A,
                            const unsigned short* __restrict__ Bt,
                            const float* __restrict__ bias,
                            unsigned short* __restrict__ C,
                            int N) {
  constexpr int K = 128;
  const int lane = threadIdx.x & 31;
  const int wave = threadIdx.x >> 5;
  const int wpb  = blockDim.x >> 5;
  const int mt   = blockIdx.x;
  const int n0   = (blockIdx.y * wpb + wave) * 64;
  if (n0 >= N) return;
  const int mrow = mt * 16 + (lane & 15);
  const int kg   = (lane >> 4) * 8;   // lanes 0-15: K[0..8)+[16..24); lanes 16-31: K[8..16)+[24..32)

  // ---- issue all fragment loads up front ----
  FragBF a[4];
  {
    const unsigned short* Ap = A + (size_t)mrow * K + kg;
    #pragma unroll
    for (int s = 0; s < 4; ++s) {
      a[s].u[0] = *(const u32x4*)(Ap + s * 32);
      a[s].u[1] = *(const u32x4*)(Ap + s * 32 + 16);
    }
  }
  FragBF b[4][4];
  #pragma unroll
  for (int t = 0; t < 4; ++t) {
    const unsigned short* Bp = Bt + (size_t)(n0 + t * 16 + (lane & 15)) * K + kg;
    #pragma unroll
    for (int s = 0; s < 4; ++s) {
      b[t][s].u[0] = *(const u32x4*)(Bp + s * 32);
      b[t][s].u[1] = *(const u32x4*)(Bp + s * 32 + 16);
    }
  }

  // ---- 16 WMMAs; earliest-loaded fragments consumed first, 4 independent accumulators ----
  v8f acc[4] = {};
  #pragma unroll
  for (int s = 0; s < 4; ++s) {
    #pragma unroll
    for (int t = 0; t < 4; ++t) {
      acc[t] = __builtin_amdgcn_wmma_f32_16x16x32_bf16(
          false, a[s].v, false, b[t][s].v, (short)0, acc[t], false, false);
    }
  }

  // ---- epilogue: bias (+relu), bf16 store ----
  #pragma unroll
  for (int t = 0; t < 4; ++t) {
    int n = n0 + t * 16 + (lane & 15);
    float bn = bias[n];
    #pragma unroll
    for (int r = 0; r < 8; ++r) {
      int row = mt * 16 + r + 8 * (lane >> 4);  // C/D layout: vgpr r -> M = r + 8*(lane/16)
      float v = acc[t][r] + bn;
      if (RELU) v = fmaxf(v, 0.0f);
      C[(size_t)row * N + n] = f2bf(v);
    }
  }
}

// ---------------- per-edge matvec + scatter: agg[dst] += h[src] @ ew_e ----------------
// one wave per edge; lane handles outputs {2*lane, 2*lane+1}; ew row loads coalesced dwords.
__global__ void k_msg(const unsigned short* __restrict__ hbf,
                      const unsigned short* __restrict__ ew,
                      const int* __restrict__ src, const int* __restrict__ dst,
                      float* __restrict__ agg) {
  int e    = blockIdx.x * (blockDim.x >> 5) + (threadIdx.x >> 5);
  int lane = threadIdx.x & 31;
  int s = src[e], d = dst[e];
  float h0 = bf2f(hbf[(size_t)s * DV + lane]);
  float h1 = bf2f(hbf[(size_t)s * DV + 32 + lane]);
  const unsigned short* ewe = ew + (size_t)e * (DV * DV);
  float a0 = 0.0f, a1 = 0.0f;
  #pragma unroll 4
  for (int i = 0; i < 32; ++i) {
    float hi = __shfl(h0, i, 32);
    unsigned int p = *(const unsigned int*)(ewe + i * DV + 2 * lane);
    a0 = fmaf(hi, bf2f((unsigned short)(p & 0xffffu)), a0);
    a1 = fmaf(hi, bf2f((unsigned short)(p >> 16)), a1);
  }
  #pragma unroll 4
  for (int i = 0; i < 32; ++i) {
    float hi = __shfl(h1, i, 32);
    unsigned int p = *(const unsigned int*)(ewe + (i + 32) * DV + 2 * lane);
    a0 = fmaf(hi, bf2f((unsigned short)(p & 0xffffu)), a0);
    a1 = fmaf(hi, bf2f((unsigned short)(p >> 16)), a1);
  }
  atomicAdd(&agg[(size_t)d * DV + 2 * lane], a0);
  atomicAdd(&agg[(size_t)d * DV + 2 * lane + 1], a1);
}

// ---------------- GRU cell per node (fp32 for accuracy) ----------------
__global__ void k_gru(const float* __restrict__ agg, const float* __restrict__ bconv,
                      const float* __restrict__ Wih, const float* __restrict__ Whh,
                      const float* __restrict__ bih, const float* __restrict__ bhh,
                      float* __restrict__ hid, unsigned short* __restrict__ hbf) {
  __shared__ float xs[4][DV];
  __shared__ float hsh[4][DV];
  int ln = threadIdx.x & (DV - 1);
  int nn = threadIdx.x >> 6;
  int v  = blockIdx.x * 4 + nn;
  float x = fmaxf(agg[(size_t)v * DV + ln] + bconv[ln], 0.0f);
  float h = hid[(size_t)v * DV + ln];
  xs[nn][ln] = x; hsh[nn][ln] = h;
  __syncthreads();
  float gr = bih[ln], gz = bih[DV + ln], gn = bih[2 * DV + ln];
  float hr = bhh[ln], hz = bhh[DV + ln], hn = bhh[2 * DV + ln];
  const float* Wr = Wih + (size_t)ln * DV;
  const float* Wz = Wih + (size_t)(DV + ln) * DV;
  const float* Wn = Wih + (size_t)(2 * DV + ln) * DV;
  const float* Ur = Whh + (size_t)ln * DV;
  const float* Uz = Whh + (size_t)(DV + ln) * DV;
  const float* Un = Whh + (size_t)(2 * DV + ln) * DV;
  #pragma unroll 8
  for (int k = 0; k < DV; ++k) {
    float xk = xs[nn][k], hk = hsh[nn][k];
    gr = fmaf(xk, Wr[k], gr); gz = fmaf(xk, Wz[k], gz); gn = fmaf(xk, Wn[k], gn);
    hr = fmaf(hk, Ur[k], hr); hz = fmaf(hk, Uz[k], hz); hn = fmaf(hk, Un[k], hn);
  }
  float r  = sigmoidf(gr + hr);
  float z  = sigmoidf(gz + hz);
  float n  = tanhf(gn + r * hn);
  float ho = (1.0f - z) * n + z * h;
  hid[(size_t)v * DV + ln] = ho;
  hbf[(size_t)v * DV + ln] = f2bf(ho);
}

// ---------------- Set2Set (6 iters, 3-layer LSTM over G=40) + predict ----------------
__global__ __launch_bounds__(1024)
void k_s2s(const float* __restrict__ h, const int* __restrict__ gid,
           const float* Wih0, const float* Whh0, const float* bih0, const float* bhh0,
           const float* Wih1, const float* Whh1, const float* bih1, const float* bhh1,
           const float* Wih2, const float* Whh2, const float* bih2, const float* bhh2,
           const float* Wp1, const float* bp1, const float* Wp2, const float* bp2,
           float* __restrict__ out,
           float* __restrict__ hs, float* __restrict__ cs, float* __restrict__ qstar,
           float* __restrict__ gates, float* __restrict__ ev,
           unsigned int* __restrict__ mkey, float* __restrict__ ssum, float* __restrict__ ro) {
  const int tid = threadIdx.x;
  const int NT  = blockDim.x;  // 1024
  const float* Wih[3] = {Wih0, Wih1, Wih2};
  const float* Whh[3] = {Whh0, Whh1, Whh2};
  const float* bih[3] = {bih0, bih1, bih2};
  const float* bhh[3] = {bhh0, bhh1, bhh2};

  for (int i = tid; i < 3 * GN * DV; i += NT) { hs[i] = 0.0f; cs[i] = 0.0f; }
  for (int i = tid; i < GN * 2 * DV; i += NT) qstar[i] = 0.0f;
  __syncthreads();

  for (int iter = 0; iter < S2S_ITERS; ++iter) {
    // --- 3-layer LSTM on q_star ---
    for (int l = 0; l < 3; ++l) {
      const float* x  = (l == 0) ? qstar : (hs + (size_t)(l - 1) * GN * DV);
      const int xdim  = (l == 0) ? 2 * DV : DV;
      const float* hl = hs + (size_t)l * GN * DV;
      for (int idx = tid; idx < GN * 4 * DV; idx += NT) {
        int g = idx >> 8, j = idx & 255;
        float acc = bih[l][j] + bhh[l][j];
        const float* Wj = Wih[l] + (size_t)j * xdim;
        const float* xg = x + (size_t)g * xdim;
        for (int k = 0; k < xdim; ++k) acc = fmaf(xg[k], Wj[k], acc);
        const float* Uj = Whh[l] + (size_t)j * DV;
        const float* hg = hl + (size_t)g * DV;
        for (int k = 0; k < DV; ++k) acc = fmaf(hg[k], Uj[k], acc);
        gates[idx] = acc;
      }
      __syncthreads();
      float* hlw = hs + (size_t)l * GN * DV;
      float* clw = cs + (size_t)l * GN * DV;
      for (int idx = tid; idx < GN * DV; idx += NT) {
        int g = idx >> 6, d = idx & 63;
        float gi = gates[g * 256 + d];
        float gf = gates[g * 256 + 64 + d];
        float gg = gates[g * 256 + 128 + d];
        float go = gates[g * 256 + 192 + d];
        float c  = clw[idx];
        float cn = sigmoidf(gf) * c + sigmoidf(gi) * tanhf(gg);
        clw[idx] = cn;
        hlw[idx] = sigmoidf(go) * tanhf(cn);
      }
      __syncthreads();
    }
    const float* q = hs + (size_t)2 * GN * DV;
    // --- attention over nodes ---
    if (tid < GN) { mkey[tid] = 0u; ssum[tid] = 0.0f; }
    for (int i = tid; i < GN * DV; i += NT) ro[i] = 0.0f;
    __syncthreads();
    for (int v = tid; v < VN; v += NT) {
      const float* hv = h + (size_t)v * DV;
      const float* qg = q + (size_t)gid[v] * DV;
      float e = 0.0f;
      for (int d = 0; d < DV; ++d) e = fmaf(hv[d], qg[d], e);
      ev[v] = e;
      unsigned int u   = __float_as_uint(e);
      unsigned int key = (u & 0x80000000u) ? ~u : (u | 0x80000000u);  // order-preserving
      atomicMax(&mkey[gid[v]], key);
    }
    __syncthreads();
    for (int v = tid; v < VN; v += NT) {
      unsigned int key = mkey[gid[v]];
      unsigned int u   = (key & 0x80000000u) ? (key & 0x7fffffffu) : ~key;
      float a = expf(ev[v] - __uint_as_float(u));
      ev[v] = a;
      atomicAdd(&ssum[gid[v]], a);
    }
    __syncthreads();
    for (int v = tid; v < VN; v += NT) {
      float alpha = ev[v] / ssum[gid[v]];
      const float* hv = h + (size_t)v * DV;
      float* rg = ro + (size_t)gid[v] * DV;
      for (int d = 0; d < DV; ++d) atomicAdd(&rg[d], alpha * hv[d]);
    }
    __syncthreads();
    for (int i = tid; i < GN * DV; i += NT) {
      int g = i >> 6, d = i & 63;
      qstar[g * 2 * DV + d]      = q[i];
      qstar[g * 2 * DV + DV + d] = ro[i];
    }
    __syncthreads();
  }
  // --- predict head ---
  if (tid < GN) {
    int g = tid;
    float acc = bp2[0];
    const float* qs = qstar + (size_t)g * 2 * DV;
    for (int j = 0; j < DV; ++j) {
      float t = bp1[j];
      for (int k = 0; k < 2 * DV; ++k) t = fmaf(qs[k], Wp1[k * DV + j], t);
      acc = fmaf(fmaxf(t, 0.0f), Wp2[j], acc);
    }
    out[g] = acc;
  }
}

extern "C" void kernel_launch(void* const* d_in, const int* in_sizes, int n_in,
                              void* d_out, int out_size, void* d_ws, size_t ws_size,
                              hipStream_t stream) {
  (void)in_sizes; (void)n_in; (void)out_size; (void)ws_size;
  const float* node_feats = (const float*)d_in[0];
  const float* edge_feats = (const float*)d_in[1];
  const int*   src        = (const int*)d_in[2];
  const int*   dst        = (const int*)d_in[3];
  const int*   gid        = (const int*)d_in[4];
  const float* W_proj = (const float*)d_in[5];
  const float* b_proj = (const float*)d_in[6];
  const float* W_e1   = (const float*)d_in[7];
  const float* b_e1   = (const float*)d_in[8];
  const float* W_e2   = (const float*)d_in[9];
  const float* b_e2   = (const float*)d_in[10];
  const float* b_conv = (const float*)d_in[11];
  const float* W_ih   = (const float*)d_in[12];
  const float* W_hh   = (const float*)d_in[13];
  const float* b_ih   = (const float*)d_in[14];
  const float* b_hh   = (const float*)d_in[15];
  const float* Wih0 = (const float*)d_in[16]; const float* Whh0 = (const float*)d_in[17];
  const float* bih0 = (const float*)d_in[18]; const float* bhh0 = (const float*)d_in[19];
  const float* Wih1 = (const float*)d_in[20]; const float* Whh1 = (const float*)d_in[21];
  const float* bih1 = (const float*)d_in[22]; const float* bhh1 = (const float*)d_in[23];
  const float* Wih2 = (const float*)d_in[24]; const float* Whh2 = (const float*)d_in[25];
  const float* bih2 = (const float*)d_in[26]; const float* bhh2 = (const float*)d_in[27];
  const float* W_p1 = (const float*)d_in[28]; const float* b_p1 = (const float*)d_in[29];
  const float* W_p2 = (const float*)d_in[30]; const float* b_p2 = (const float*)d_in[31];
  float* out = (float*)d_out;

  // workspace carve-up (256B aligned)
  char* ws = (char*)d_ws;
  size_t off = 0;
  auto alloc = [&](size_t bytes) -> char* {
    char* p = ws + off;
    off = (off + bytes + 255) & ~(size_t)255;
    return p;
  };
  float*          hid  = (float*)alloc((size_t)VN * DV * 4);
  unsigned short* hbf  = (unsigned short*)alloc((size_t)VN * DV * 2);
  unsigned short* efb  = (unsigned short*)alloc((size_t)EN * EDGE_IN * 2);
  unsigned short* w1t  = (unsigned short*)alloc((size_t)EDGE_IN * EDGE_HID * 2);
  unsigned short* w2t  = (unsigned short*)alloc((size_t)EDGE_HID * DV * DV * 2);
  unsigned short* zb   = (unsigned short*)alloc((size_t)EN * EDGE_HID * 2);
  unsigned short* ewb  = (unsigned short*)alloc((size_t)EN * DV * DV * 2);   // 327 MB bf16
  float*          agg  = (float*)alloc((size_t)VN * DV * 4);
  float*          hsb  = (float*)alloc((size_t)3 * GN * DV * 4);
  float*          csb  = (float*)alloc((size_t)3 * GN * DV * 4);
  float*          qst  = (float*)alloc((size_t)GN * 2 * DV * 4);
  float*          gts  = (float*)alloc((size_t)GN * 4 * DV * 4);
  float*          evb  = (float*)alloc((size_t)VN * 4);
  unsigned int*   mk   = (unsigned int*)alloc((size_t)GN * 4);
  float*          ssm  = (float*)alloc((size_t)GN * 4);
  float*          rob  = (float*)alloc((size_t)GN * DV * 4);

  // precision conversions (once)
  { int n = EN * EDGE_IN;        k_cvt_bf16<<<(n + 255) / 256, 256, 0, stream>>>(edge_feats, efb, n); }
  { int n = EDGE_IN * EDGE_HID;  k_cvt_transpose<<<(n + 255) / 256, 256, 0, stream>>>(W_e1, w1t, EDGE_IN, EDGE_HID); }
  { int n = EDGE_HID * DV * DV;  k_cvt_transpose<<<(n + 255) / 256, 256, 0, stream>>>(W_e2, w2t, EDGE_HID, DV * DV); }

  // node projection
  k_proj<<<VN, DV, 0, stream>>>(node_feats, W_proj, b_proj, hid, hbf);

  // edge network: z = relu(ef @ W_e1 + b_e1); ew = z @ W_e2 + b_e2  (bf16 WMMA, K=128)
  k_gemm_bf16<1><<<dim3(EN / 16, 1), 64, 0, stream>>>(efb, w1t, b_e1, zb, EDGE_HID);
  k_gemm_bf16<0><<<dim3(EN / 16, 16), 128, 0, stream>>>(zb, w2t, b_e2, ewb, DV * DV);

  // message passing
  for (int s = 0; s < NSTEPS; ++s) {
    { int n = VN * DV; k_fill0<<<(n + 255) / 256, 256, 0, stream>>>(agg, n); }
    k_msg<<<EN / 8, 256, 0, stream>>>(hbf, ewb, src, dst, agg);
    k_gru<<<VN / 4, 256, 0, stream>>>(agg, b_conv, W_ih, W_hh, b_ih, b_hh, hid, hbf);
  }

  // Set2Set readout + predict
  k_s2s<<<1, 1024, 0, stream>>>(hid, gid,
      Wih0, Whh0, bih0, bhh0, Wih1, Whh1, bih1, bhh1, Wih2, Whh2, bih2, bhh2,
      W_p1, b_p1, W_p2, b_p2, out,
      hsb, csb, qst, gts, evb, mk, ssm, rob);
}
